// SpanModel_7610682049131
// MI455X (gfx1250) — matrix operand order repaired and checked by reading, not compiled, not run on hardware
//
#include <hip/hip_runtime.h>
#include <hip/hip_bf16.h>

// ---------------- problem constants ----------------
#define BB     8
#define SSEQ   512
#define DD     768
#define CC     25
#define TT     384
#define KSPAN  12
#define KNEG   4
#define PROMPT 50          // 2*C prompt region offset
#define HH     3072        // 4*D
#define NSPAN  (TT*KSPAN)  // 4608
#define CPAD   32          // C padded to a WMMA tile in N
#define SELPAD 112         // 100 selected rows padded to multiple of 16
#define MW     (BB*TT)     // 3072 word rows
#define MP     208         // B*C=200 padded to multiple of 16

typedef __attribute__((ext_vector_type(16))) __bf16 v16bf;
typedef __attribute__((ext_vector_type(8)))  float  v8f;

__device__ __forceinline__ __bf16 f2bf(float f) {
  union { float f; unsigned u; } v; v.f = f;
  unsigned r = v.u + 0x7FFFu + ((v.u >> 16) & 1u);   // round-to-nearest-even
  unsigned short h = (unsigned short)(r >> 16);
  return __builtin_bit_cast(__bf16, h);
}

union FragU { uint4 q[2]; v16bf v; };

__device__ __forceinline__ v16bf load_fragA(const __bf16* p) {
  // lanes<16 hold K = 0..15 contiguous (row ptr already includes half*16)
  FragU u;
  u.q[0] = *(const uint4*)(p);
  u.q[1] = *(const uint4*)(p + 8);
  return u.v;
}
__device__ __forceinline__ v16bf load_fragB(const __bf16* p) {
  // lanes<16 hold K = 0..15 contiguous (row ptr already includes half*16)
  FragU u;
  u.q[0] = *(const uint4*)(p);
  u.q[1] = *(const uint4*)(p + 8);
  return u.v;
}

// ---------------------------------------------------------------------------
// Generic batched GEMM (1 wave = one 16x16 tile). Used for odd shapes.
// C[b] (f32 and/or bf16, MxN) = act(A[b](MxK) @ Bt[b](NxK)^T)
// ---------------------------------------------------------------------------
__global__ void wmma_gemm_bf16(const __bf16* __restrict__ A,
                               const __bf16* __restrict__ Bt,
                               float* __restrict__ Cf,
                               __bf16* __restrict__ Cbf,
                               int M, int N, int Kd,
                               unsigned long long sA, unsigned long long sB,
                               unsigned long long sC,
                               int tilesM, int tilesN, int nBatch, int relu)
{
  int wv   = blockIdx.x * (blockDim.x >> 5) + (threadIdx.x >> 5);
  int tper = tilesM * tilesN;
  if (wv >= tper * nBatch) return;
  int lane = threadIdx.x & 31;
  int b    = wv / tper;
  int t    = wv - b * tper;
  int tm   = (t / tilesN) << 4;
  int tn   = (t - (t / tilesN) * tilesN) << 4;
  int half = lane >> 4;
  int l15  = lane & 15;

  const __bf16* arow = A + (size_t)b * sA + (size_t)(tm + l15) * Kd + half * 16;
  const __bf16* brow = Bt + (size_t)b * sB + (size_t)(tn + l15) * Kd + half * 16;

  v8f acc = {};
  for (int k0 = 0; k0 < Kd; k0 += 32) {
    v16bf af = load_fragA(arow + k0);
    v16bf bf = load_fragB(brow + k0);
    acc = __builtin_amdgcn_wmma_f32_16x16x32_bf16(false, af, false, bf,
                                                  (short)0, acc, false, false);
  }

  size_t cbase = (size_t)b * sC + (size_t)(tm + half * 8) * N + (size_t)(tn + l15);
  #pragma unroll
  for (int r = 0; r < 8; ++r) {
    float v = acc[r];
    if (relu) v = v > 0.f ? v : 0.f;
    if (Cf)  Cf [cbase + (size_t)r * N] = v;
    if (Cbf) Cbf[cbase + (size_t)r * N] = f2bf(v);
  }
}

// ---------------------------------------------------------------------------
// Register-blocked GEMM: 1 wave = 32x32 output (2x2 WMMA tiles).
// 4 WMMAs per 8 fragment loads -> 2x the L2 arithmetic intensity of the
// generic kernel. Requires M%32==0, N%32==0. Prefetches next k-panel
// (lowers to global_prefetch_b8 on gfx1250).
// ---------------------------------------------------------------------------
__global__ void wmma_gemm_bf16_2x2(const __bf16* __restrict__ A,
                                   const __bf16* __restrict__ Bt,
                                   float* __restrict__ Cf,
                                   __bf16* __restrict__ Cbf,
                                   int M, int N, int Kd, int tilesN2, int relu)
{
  int wv = blockIdx.x * (blockDim.x >> 5) + (threadIdx.x >> 5);
  int totalTiles = (M >> 5) * tilesN2;
  if (wv >= totalTiles) return;
  int lane = threadIdx.x & 31;
  int tm = (wv / tilesN2) << 5;
  int tn = (wv - (wv / tilesN2) * tilesN2) << 5;
  int half = lane >> 4;
  int l15  = lane & 15;

  const __bf16* arow0 = A + (size_t)(tm + l15) * Kd + half * 16;
  const __bf16* arow1 = arow0 + (size_t)16 * Kd;
  const __bf16* brow0 = Bt + (size_t)(tn + l15) * Kd + half * 16;
  const __bf16* brow1 = brow0 + (size_t)16 * Kd;

  v8f acc00 = {}, acc01 = {}, acc10 = {}, acc11 = {};
  for (int k0 = 0; k0 < Kd; k0 += 32) {
    if (k0 + 256 < Kd) {            // prefetch next panels into cache
      __builtin_prefetch(arow0 + k0 + 256, 0, 3);
      __builtin_prefetch(arow1 + k0 + 256, 0, 3);
      __builtin_prefetch(brow0 + k0 + 256, 0, 3);
      __builtin_prefetch(brow1 + k0 + 256, 0, 3);
    }
    v16bf af0 = load_fragA(arow0 + k0);
    v16bf af1 = load_fragA(arow1 + k0);
    v16bf bf0 = load_fragB(brow0 + k0);
    v16bf bf1 = load_fragB(brow1 + k0);
    acc00 = __builtin_amdgcn_wmma_f32_16x16x32_bf16(false, af0, false, bf0,
                                                    (short)0, acc00, false, false);
    acc01 = __builtin_amdgcn_wmma_f32_16x16x32_bf16(false, af0, false, bf1,
                                                    (short)0, acc01, false, false);
    acc10 = __builtin_amdgcn_wmma_f32_16x16x32_bf16(false, af1, false, bf0,
                                                    (short)0, acc10, false, false);
    acc11 = __builtin_amdgcn_wmma_f32_16x16x32_bf16(false, af1, false, bf1,
                                                    (short)0, acc11, false, false);
  }

  v8f* accs[4] = { &acc00, &acc01, &acc10, &acc11 };
  #pragma unroll
  for (int ti = 0; ti < 2; ++ti) {
    #pragma unroll
    for (int tj = 0; tj < 2; ++tj) {
      v8f a = *accs[ti * 2 + tj];
      size_t cbase = (size_t)(tm + ti * 16 + half * 8) * N + (size_t)(tn + tj * 16 + l15);
      #pragma unroll
      for (int r = 0; r < 8; ++r) {
        float v = a[r];
        if (relu) v = v > 0.f ? v : 0.f;
        if (Cf)  Cf [cbase + (size_t)r * N] = v;
        if (Cbf) Cbf[cbase + (size_t)r * N] = f2bf(v);
      }
    }
  }
}

// ---------------- elementwise / data-movement kernels ----------------

__global__ void extract_words(const float* __restrict__ emb,
                              const int* __restrict__ tlen,
                              __bf16* __restrict__ wordsBF)
{
  int tid = blockIdx.x * blockDim.x + threadIdx.x;
  if (tid >= MW * DD) return;
  int d = tid % DD;
  int bt = tid / DD;
  int t = bt % TT, b = bt / TT;
  float v = (t < tlen[b]) ? emb[((size_t)(b * SSEQ + PROMPT + t)) * DD + d] : 0.f;
  wordsBF[tid] = f2bf(v);
}

__global__ void extract_prompts(const float* __restrict__ emb,
                                __bf16* __restrict__ p0)
{
  int tid = blockIdx.x * blockDim.x + threadIdx.x;
  if (tid >= MP * DD) return;
  int d = tid % DD, r = tid / DD;
  float v = 0.f;
  if (r < BB * CC) {
    int b = r / CC, c = r % CC;
    v = emb[((size_t)(b * SSEQ + 2 * c)) * DD + d];
  }
  p0[tid] = f2bf(v);
}

__global__ void convT(const float* __restrict__ W, __bf16* __restrict__ Wt,
                      int rows, int cols)
{
  int tid = blockIdx.x * blockDim.x + threadIdx.x;
  if (tid >= rows * cols) return;
  int c = tid % cols, r = tid / cols;
  Wt[(size_t)c * rows + r] = f2bf(W[(size_t)r * cols + c]);
}

__global__ void pad_prompts(const float* __restrict__ Pf, __bf16* __restrict__ pp)
{
  int tid = blockIdx.x * blockDim.x + threadIdx.x;
  if (tid >= BB * CPAD * DD) return;
  int d = tid % DD;
  int bc = tid / DD;
  int c = bc % CPAD, b = bc / CPAD;
  float v = (c < CC) ? Pf[((size_t)(b * CC + c)) * DD + d] : 0.f;
  pp[tid] = f2bf(v);
}

__global__ void combine_scores(const float* __restrict__ As,
                               const float* __restrict__ Bs,
                               const int* __restrict__ span,
                               const int* __restrict__ mask,
                               float* __restrict__ out)
{
  int tid = blockIdx.x * blockDim.x + threadIdx.x;
  if (tid >= BB * NSPAN * CC) return;
  int c = tid % CC;
  int bi = tid / CC;
  int i = bi % NSPAN, b = bi / NSPAN;
  int m  = mask[b * NSPAN + i];
  int st = span[((size_t)(b * NSPAN + i)) * 2 + 0] * m;
  int en = span[((size_t)(b * NSPAN + i)) * 2 + 1] * m;
  out[tid] = As[((size_t)(b * TT + st)) * CPAD + c]
           + Bs[((size_t)(b * TT + en)) * CPAD + c];
}

__global__ void topk_kernel(const float* __restrict__ sc, int* __restrict__ out)
{
  int tid = blockIdx.x * blockDim.x + threadIdx.x;
  if (tid >= BB * CC) return;
  int b = tid / CC, c = tid % CC;
  float v0 = -3.4e38f, v1 = v0, v2 = v0, v3 = v0;
  int i0 = 0, i1 = 0, i2 = 0, i3 = 0;
  const float* p = sc + (size_t)b * NSPAN * CC + c;
  for (int i = 0; i < NSPAN; ++i) {
    float x = p[(size_t)i * CC];
    if (x > v3) {
      if (x > v0)      { v3=v2;i3=i2; v2=v1;i2=i1; v1=v0;i1=i0; v0=x;i0=i; }
      else if (x > v1) { v3=v2;i3=i2; v2=v1;i2=i1; v1=x;i1=i; }
      else if (x > v2) { v3=v2;i3=i2; v2=x;i2=i; }
      else             { v3=x;i3=i; }
    }
  }
  int* q = out + (size_t)(b * CC + c) * KNEG;
  q[0] = i0; q[1] = i1; q[2] = i2; q[3] = i3;
}

__global__ void select_kernel(const float* __restrict__ Af,
                              const float* __restrict__ Bf,
                              const int* __restrict__ topk,
                              const int* __restrict__ span,
                              const int* __restrict__ mask,
                              __bf16* __restrict__ sel)
{
  int tid = blockIdx.x * blockDim.x + threadIdx.x;
  if (tid >= BB * SELPAD * DD) return;
  int d = tid % DD;
  int rj = tid / DD;
  int j = rj % SELPAD, b = rj / SELPAD;
  float v = 0.f;
  if (j < KNEG * CC) {
    int kneg = j / CC, c = j % CC;            // selection order: kneg*C + c
    int i = topk[(size_t)(b * CC + c) * KNEG + kneg];
    int m = mask[b * NSPAN + i];
    int st = span[((size_t)(b * NSPAN + i)) * 2 + 0] * m;
    int en = span[((size_t)(b * NSPAN + i)) * 2 + 1] * m;
    v = Af[((size_t)(b * TT + st)) * DD + d]
      + Bf[((size_t)(b * TT + en)) * DD + d];
  }
  sel[tid] = f2bf(v);
}

__global__ void combine_so(const float* __restrict__ spS,
                           const float* __restrict__ opS,
                           float* __restrict__ out)
{
  int tid = blockIdx.x * blockDim.x + threadIdx.x;
  if (tid >= BB * 100 * 100 * CC) return;
  int c = tid % CC;
  int r = tid / CC;
  int o = r % 100; r /= 100;
  int s = r % 100; int b = r / 100;
  out[tid] = spS[((size_t)(b * SELPAD + s)) * CPAD + c]
           + opS[((size_t)(b * SELPAD + o)) * CPAD + c];
}

// ---------------- host side ----------------

static inline int cdiv(long long a, long long b) { return (int)((a + b - 1) / b); }

static void gemm(hipStream_t s, const __bf16* A, const __bf16* Bt,
                 float* Cf, __bf16* Cbf, int M, int N, int Kd,
                 long long sA, long long sB, long long sC, int nB, int relu)
{
  int tm = M / 16, tn = N / 16;
  long long waves = (long long)tm * tn * nB;
  wmma_gemm_bf16<<<cdiv(waves, 8), 256, 0, s>>>(
      A, Bt, Cf, Cbf, M, N, Kd,
      (unsigned long long)sA, (unsigned long long)sB, (unsigned long long)sC,
      tm, tn, nB, relu);
}

static void gemm_blk(hipStream_t s, const __bf16* A, const __bf16* Bt,
                     float* Cf, __bf16* Cbf, int M, int N, int Kd, int relu)
{
  int tn2 = N / 32;
  long long waves = (long long)(M / 32) * tn2;
  wmma_gemm_bf16_2x2<<<cdiv(waves, 8), 256, 0, s>>>(A, Bt, Cf, Cbf, M, N, Kd,
                                                    tn2, relu);
}

extern "C" void kernel_launch(void* const* d_in, const int* in_sizes, int n_in,
                              void* d_out, int out_size, void* d_ws, size_t ws_size,
                              hipStream_t stream)
{
  (void)in_sizes; (void)n_in; (void)out_size; (void)ws_size;
  const float* emb    = (const float*)d_in[0];
  const int*   tlen   = (const int*)d_in[3];
  const int*   span   = (const int*)d_in[4];
  const int*   msub   = (const int*)d_in[5];
  const int*   mob    = (const int*)d_in[6];
  const float* Ws_sub = (const float*)d_in[7];
  const float* We_sub = (const float*)d_in[8];
  const float* Wo_sub = (const float*)d_in[9];
  const float* Ws_ob  = (const float*)d_in[10];
  const float* We_ob  = (const float*)d_in[11];
  const float* Wo_ob  = (const float*)d_in[12];
  const float* Wso    = (const float*)d_in[13];
  const float* Wp1    = (const float*)d_in[14];
  const float* Wp2    = (const float*)d_in[15];

  float* out_sub = (float*)d_out;                       // [B,T,K,C]
  float* out_ob  = out_sub + (size_t)BB * NSPAN * CC;
  float* out_so  = out_ob  + (size_t)BB * NSPAN * CC;   // [B,100,100,C]

  char* p = (char*)d_ws;
  auto alloc = [&](size_t bytes) -> void* {
    void* r = (void*)p; p += (bytes + 255) & ~(size_t)255; return r;
  };

  __bf16* wordsBF = (__bf16*)alloc((size_t)MW * DD * 2);
  __bf16* p0bf    = (__bf16*)alloc((size_t)MP * DD * 2);
  __bf16* hid     = (__bf16*)alloc((size_t)MP * HH * 2);
  float*  Pf      = (float*) alloc((size_t)MP * DD * 4);
  __bf16* pPad    = (__bf16*)alloc((size_t)BB * CPAD * DD * 2);

  __bf16* WsT_sub = (__bf16*)alloc((size_t)DD * DD * 2);
  __bf16* WeT_sub = (__bf16*)alloc((size_t)DD * DD * 2);
  __bf16* Wo1T_sub= (__bf16*)alloc((size_t)DD * DD * 2);
  __bf16* Wo2T_sub= (__bf16*)alloc((size_t)DD * DD * 2);
  __bf16* WsT_ob  = (__bf16*)alloc((size_t)DD * DD * 2);
  __bf16* WeT_ob  = (__bf16*)alloc((size_t)DD * DD * 2);
  __bf16* Wo1T_ob = (__bf16*)alloc((size_t)DD * DD * 2);
  __bf16* Wo2T_ob = (__bf16*)alloc((size_t)DD * DD * 2);
  __bf16* Wso1T   = (__bf16*)alloc((size_t)DD * DD * 2);
  __bf16* Wso2T   = (__bf16*)alloc((size_t)DD * DD * 2);
  __bf16* Wp1T    = (__bf16*)alloc((size_t)DD * HH * 2);
  __bf16* Wp2T    = (__bf16*)alloc((size_t)HH * DD * 2);

  __bf16* tmp_bf  = (__bf16*)alloc((size_t)MW * DD * 2);
  __bf16* tmp2_bf = (__bf16*)alloc((size_t)MW * DD * 2);
  float*  A_sub   = (float*) alloc((size_t)MW * DD * 4);
  float*  B_sub   = (float*) alloc((size_t)MW * DD * 4);
  float*  A_ob    = (float*) alloc((size_t)MW * DD * 4);
  float*  B_ob    = (float*) alloc((size_t)MW * DD * 4);
  float*  As_sub  = (float*) alloc((size_t)BB * TT * CPAD * 4);
  float*  Bs_sub  = (float*) alloc((size_t)BB * TT * CPAD * 4);
  float*  As_ob   = (float*) alloc((size_t)BB * TT * CPAD * 4);
  float*  Bs_ob   = (float*) alloc((size_t)BB * TT * CPAD * 4);
  int*    idx_sub = (int*)   alloc((size_t)BB * CC * KNEG * 4);
  int*    idx_ob  = (int*)   alloc((size_t)BB * CC * KNEG * 4);
  __bf16* sel_sub = (__bf16*)alloc((size_t)BB * SELPAD * DD * 2);
  __bf16* sel_ob  = (__bf16*)alloc((size_t)BB * SELPAD * DD * 2);
  __bf16* sp_bf   = (__bf16*)alloc((size_t)BB * SELPAD * DD * 2);
  __bf16* op_bf   = (__bf16*)alloc((size_t)BB * SELPAD * DD * 2);
  float*  spS     = (float*) alloc((size_t)BB * SELPAD * CPAD * 4);
  float*  opS     = (float*) alloc((size_t)BB * SELPAD * CPAD * 4);

  const int TPB = 256;

  // ---- extraction + weight conversion ----
  extract_words<<<cdiv((long long)MW * DD, TPB), TPB, 0, stream>>>(emb, tlen, wordsBF);
  extract_prompts<<<cdiv((long long)MP * DD, TPB), TPB, 0, stream>>>(emb, p0bf);
  convT<<<cdiv((long long)DD * DD, TPB), TPB, 0, stream>>>(Ws_sub, WsT_sub, DD, DD);
  convT<<<cdiv((long long)DD * DD, TPB), TPB, 0, stream>>>(We_sub, WeT_sub, DD, DD);
  convT<<<cdiv((long long)DD * DD, TPB), TPB, 0, stream>>>(Wo_sub,            Wo1T_sub, DD, DD);
  convT<<<cdiv((long long)DD * DD, TPB), TPB, 0, stream>>>(Wo_sub + DD * DD,  Wo2T_sub, DD, DD);
  convT<<<cdiv((long long)DD * DD, TPB), TPB, 0, stream>>>(Ws_ob, WsT_ob, DD, DD);
  convT<<<cdiv((long long)DD * DD, TPB), TPB, 0, stream>>>(We_ob, WeT_ob, DD, DD);
  convT<<<cdiv((long long)DD * DD, TPB), TPB, 0, stream>>>(Wo_ob,            Wo1T_ob, DD, DD);
  convT<<<cdiv((long long)DD * DD, TPB), TPB, 0, stream>>>(Wo_ob + DD * DD,  Wo2T_ob, DD, DD);
  convT<<<cdiv((long long)DD * DD, TPB), TPB, 0, stream>>>(Wso,            Wso1T, DD, DD);
  convT<<<cdiv((long long)DD * DD, TPB), TPB, 0, stream>>>(Wso + DD * DD,  Wso2T, DD, DD);
  convT<<<cdiv((long long)DD * HH, TPB), TPB, 0, stream>>>(Wp1, Wp1T, DD, HH);
  convT<<<cdiv((long long)HH * DD, TPB), TPB, 0, stream>>>(Wp2, Wp2T, HH, DD);

  // ---- prompt MLP: P = relu(P0 @ Wp1) @ Wp2 ----  (M=208: generic kernel)
  gemm(stream, p0bf, Wp1T, nullptr, hid, MP, HH, DD, 0, 0, 0, 1, 1);
  gemm(stream, hid, Wp2T, Pf, nullptr, MP, DD, HH, 0, 0, 0, 1, 0);
  pad_prompts<<<cdiv((long long)BB * CPAD * DD, TPB), TPB, 0, stream>>>(Pf, pPad);

  // ---- span branch (sub): register-blocked WMMA for the big GEMMs ----
  gemm_blk(stream, wordsBF, WsT_sub, nullptr, tmp_bf, MW, DD, DD, 1);
  gemm_blk(stream, tmp_bf, Wo1T_sub, A_sub, tmp2_bf, MW, DD, DD, 0);
  gemm(stream, tmp2_bf, pPad, As_sub, nullptr, TT, CPAD, DD,
       (long long)TT * DD, (long long)CPAD * DD, (long long)TT * CPAD, BB, 0);
  gemm_blk(stream, wordsBF, WeT_sub, nullptr, tmp_bf, MW, DD, DD, 1);
  gemm_blk(stream, tmp_bf, Wo2T_sub, B_sub, tmp2_bf, MW, DD, DD, 0);
  gemm(stream, tmp2_bf, pPad, Bs_sub, nullptr, TT, CPAD, DD,
       (long long)TT * DD, (long long)CPAD * DD, (long long)TT * CPAD, BB, 0);
  combine_scores<<<cdiv((long long)BB * NSPAN * CC, TPB), TPB, 0, stream>>>(
      As_sub, Bs_sub, span, msub, out_sub);

  // ---- span branch (ob) ----
  gemm_blk(stream, wordsBF, WsT_ob, nullptr, tmp_bf, MW, DD, DD, 1);
  gemm_blk(stream, tmp_bf, Wo1T_ob, A_ob, tmp2_bf, MW, DD, DD, 0);
  gemm(stream, tmp2_bf, pPad, As_ob, nullptr, TT, CPAD, DD,
       (long long)TT * DD, (long long)CPAD * DD, (long long)TT * CPAD, BB, 0);
  gemm_blk(stream, wordsBF, WeT_ob, nullptr, tmp_bf, MW, DD, DD, 1);
  gemm_blk(stream, tmp_bf, Wo2T_ob, B_ob, tmp2_bf, MW, DD, DD, 0);
  gemm(stream, tmp2_bf, pPad, Bs_ob, nullptr, TT, CPAD, DD,
       (long long)TT * DD, (long long)CPAD * DD, (long long)TT * CPAD, BB, 0);
  combine_scores<<<cdiv((long long)BB * NSPAN * CC, TPB), TPB, 0, stream>>>(
      As_ob, Bs_ob, span, mob, out_ob);

  // ---- top-k negatives + selection ----
  topk_kernel<<<cdiv(BB * CC, 64), 64, 0, stream>>>(out_sub, idx_sub);
  topk_kernel<<<cdiv(BB * CC, 64), 64, 0, stream>>>(out_ob, idx_ob);
  select_kernel<<<cdiv((long long)BB * SELPAD * DD, TPB), TPB, 0, stream>>>(
      A_sub, B_sub, idx_sub, span, msub, sel_sub);
  select_kernel<<<cdiv((long long)BB * SELPAD * DD, TPB), TPB, 0, stream>>>(
      A_ob, B_ob, idx_ob, span, mob, sel_ob);

  // ---- so-rep projections + scoring ----  (M=896 is 32-aligned: blocked)
  gemm_blk(stream, sel_sub, Wso1T, nullptr, sp_bf, BB * SELPAD, DD, DD, 0);
  gemm_blk(stream, sel_ob,  Wso2T, nullptr, op_bf, BB * SELPAD, DD, DD, 0);
  gemm(stream, sp_bf, pPad, spS, nullptr, SELPAD, CPAD, DD,
       (long long)SELPAD * DD, (long long)CPAD * DD, (long long)SELPAD * CPAD, BB, 0);
  gemm(stream, op_bf, pPad, opS, nullptr, SELPAD, CPAD, DD,
       (long long)SELPAD * DD, (long long)CPAD * DD, (long long)SELPAD * CPAD, BB, 0);
  combine_so<<<cdiv((long long)BB * 100 * 100 * CC, TPB), TPB, 0, stream>>>(
      spS, opS, out_so);
}
